// UCL_13331578487560
// MI455X (gfx1250) — matrix operand order repaired and checked by reading, not compiled, not run on hardware
//
#include <hip/hip_runtime.h>
#include <hip/hip_bf16.h>

typedef __attribute__((ext_vector_type(16))) _Float16 v16h;
typedef __attribute__((ext_vector_type(8)))  _Float16 v8h;
typedef __attribute__((ext_vector_type(8)))  float    v8f;

#define NROWS 8192
#define PANEL_COLS 64
#define PANEL_BYTES (PANEL_COLS * 256)  /* 64 z-rows x 128 f16 = 16 KB, contiguous */

// ---------------------------------------------------------------------------
// WMMA fragment loaders (CDNA5 16x16x32 f16 layouts, wave32)
// ---------------------------------------------------------------------------
__device__ __forceinline__ v8f v8f_zero() {
  v8f z;
#pragma unroll
  for (int i = 0; i < 8; ++i) z[i] = 0.0f;
  return z;
}

// A matrix 16x32 f16: lane L holds row M=L%16; halves j = K + (L/16)*16 + j.
__device__ __forceinline__ v16h load_a_frag(const _Float16* __restrict__ A, int lda,
                                            int row0, int kc, int lane) {
  const int m  = lane & 15;
  const int kb = kc + ((lane >> 4) << 4);
  const _Float16* p = A + (size_t)(row0 + m) * lda + kb;
  v8h lo = *(const v8h*)p;
  v8h hi = *(const v8h*)(p + 8);
  v16h f;
#pragma unroll
  for (int j = 0; j < 8; ++j) { f[j] = lo[j]; f[j + 8] = hi[j]; }
  return f;
}

// B matrix 32x16 f16: lane L holds column N=L%16; halves j = K + (L/16)*16 + j.
// Column n of B == row n of the row-major source (B = source^T), contiguous.
__device__ __forceinline__ v16h load_b_frag(const _Float16* __restrict__ Bm, int ldb,
                                            int col0, int kc, int lane) {
  const int n  = lane & 15;
  const int kb = kc + ((lane >> 4) << 4);
  return *(const v16h*)(Bm + (size_t)(col0 + n) * ldb + kb);
}

// Same B fragment but sourced from an LDS panel (ldb = 128 halves per row).
__device__ __forceinline__ v16h lds_b_frag(const _Float16* lb, int ct2, int kc, int lane) {
  const int n  = lane & 15;
  const int kb = kc + ((lane >> 4) << 4);
  return *(const v16h*)(lb + (ct2 * 16 + n) * 128 + kb);
}

// ---------------------------------------------------------------------------
// CDNA5 async global->LDS copy (ASYNCcnt tracked) + waits
// ---------------------------------------------------------------------------
__device__ __forceinline__ void ucl_async_ld16(unsigned int lds_off, const void* gaddr) {
  asm volatile("global_load_async_to_lds_b128 %0, %1, off"
               :: "v"(lds_off), "v"(gaddr)
               : "memory");
}
__device__ __forceinline__ void ucl_wait_async4() {
  asm volatile("s_wait_asynccnt 0x4" ::: "memory");
}
__device__ __forceinline__ void ucl_wait_async0() {
  asm volatile("s_wait_asynccnt 0x0" ::: "memory");
}

// ---------------------------------------------------------------------------
// Elementwise prep kernels
// ---------------------------------------------------------------------------
__global__ void ucl_cast_pad(const float* __restrict__ X, _Float16* __restrict__ Y,
                             int Kin, int Kout, int total) {
  int idx = blockIdx.x * 256 + threadIdx.x;
  if (idx >= total) return;
  int r = idx / Kout, c = idx - r * Kout;
  Y[idx] = (c < Kin) ? (_Float16)X[(size_t)r * Kin + c] : (_Float16)0.0f;
}

// W [K,128] f32 -> WT [128,Kp] f16 (zero padded along K)
__global__ void ucl_wtrans(const float* __restrict__ Win, _Float16* __restrict__ WT,
                           int K, int Kp, int total) {
  int idx = blockIdx.x * 256 + threadIdx.x;
  if (idx >= total) return;
  int n = idx / Kp, k = idx - n * Kp;
  WT[idx] = (k < K) ? (_Float16)Win[(size_t)k * 128 + n] : (_Float16)0.0f;
}

// ---------------------------------------------------------------------------
// GEMM: out[M,128] = A[M,Kp](f16) @ WT[128,Kp]^T + bias  (f32 out)
// 256 threads = 8 waves; wave w owns rows [blk*128 + w*16, +16), all 128 cols.
// ---------------------------------------------------------------------------
__global__ void __launch_bounds__(256)
ucl_wmma_gemm_bias(const _Float16* __restrict__ A, const _Float16* __restrict__ W,
                   const float* __restrict__ bias, float* __restrict__ out, int Kp) {
  const int lane = threadIdx.x & 31;
  const int wave = threadIdx.x >> 5;
  const int row0 = blockIdx.x * 128 + wave * 16;
  v8f acc[8];
#pragma unroll
  for (int t = 0; t < 8; ++t) acc[t] = v8f_zero();
  for (int kc = 0; kc < Kp; kc += 32) {
    v16h a = load_a_frag(A, Kp, row0, kc, lane);
#pragma unroll
    for (int ct = 0; ct < 8; ++ct) {
      v16h b = load_b_frag(W, Kp, ct * 16, kc, lane);
      acc[ct] = __builtin_amdgcn_wmma_f32_16x16x32_f16(
          false, a, false, b, (short)0, acc[ct], false, false);
    }
  }
  const int rb = row0 + ((lane >> 4) << 3);
  const int n  = lane & 15;
#pragma unroll
  for (int ct = 0; ct < 8; ++ct) {
    const int col = ct * 16 + n;
    const float bv = bias[col];
#pragma unroll
    for (int i = 0; i < 8; ++i)
      out[(size_t)(rb + i) * 128 + col] = acc[ct][i] + bv;
  }
}

// ---------------------------------------------------------------------------
// BatchNorm stats: scale = g*rsqrt(var+eps), shift = bt - mu*scale
// ---------------------------------------------------------------------------
__global__ void ucl_col_stats(const float* __restrict__ H, const float* __restrict__ g,
                              const float* __restrict__ bt, float* __restrict__ scale,
                              float* __restrict__ shift) {
  const int col = blockIdx.x;
  float s = 0.0f, s2 = 0.0f;
  for (int r = threadIdx.x; r < NROWS; r += 256) {
    float v = H[(size_t)r * 128 + col];
    s += v; s2 += v * v;
  }
  __shared__ float sh[256], sh2[256];
  sh[threadIdx.x] = s; sh2[threadIdx.x] = s2;
  __syncthreads();
  for (int st = 128; st > 0; st >>= 1) {
    if (threadIdx.x < st) {
      sh[threadIdx.x]  += sh[threadIdx.x + st];
      sh2[threadIdx.x] += sh2[threadIdx.x + st];
    }
    __syncthreads();
  }
  if (threadIdx.x == 0) {
    float mu  = sh[0] * (1.0f / NROWS);
    float var = sh2[0] * (1.0f / NROWS) - mu * mu;
    float sc  = g[col] * rsqrtf(var + 1e-5f);
    scale[col] = sc;
    shift[col] = bt[col] - mu * sc;
  }
}

__global__ void ucl_bn_relu(const float* __restrict__ H, const float* __restrict__ scale,
                            const float* __restrict__ shift, _Float16* __restrict__ HN) {
  int idx = blockIdx.x * 256 + threadIdx.x;  // total = 8192*128
  int col = idx & 127;
  float v = H[idx] * scale[col] + shift[col];
  HN[idx] = (_Float16)fmaxf(v, 0.0f);
}

// One warp per row: L2 normalize, write f16
__global__ void ucl_row_norm(const float* __restrict__ ZT, _Float16* __restrict__ Z) {
  const int row  = blockIdx.x * 8 + (threadIdx.x >> 5);
  const int lane = threadIdx.x & 31;
  const float* p = ZT + (size_t)row * 128;
  float v[4]; float ss = 0.0f;
#pragma unroll
  for (int j = 0; j < 4; ++j) { v[j] = p[lane + 32 * j]; ss += v[j] * v[j]; }
#pragma unroll
  for (int m = 1; m < 32; m <<= 1) ss += __shfl_xor(ss, m);
  float inv = 1.0f / fmaxf(sqrtf(ss), 1e-12f);
#pragma unroll
  for (int j = 0; j < 4; ++j)
    Z[(size_t)row * 128 + lane + 32 * j] = (_Float16)(v[j] * inv);
}

// ---------------------------------------------------------------------------
// Fused GEMM + shifted-sum-of-exp: S[r] = sum_c exp(2*(za[r].zb[c]) - 2)
// (dot in [-1,1] -> fixed shift 2 is a safe global max; lse = log(S)+2 later).
// zb panels (64 rows = 16 KB contiguous) are double-buffered in LDS via
// global_load_async_to_lds_b128 and shared by all 8 waves of the block,
// cutting L2 read traffic 8x vs per-wave streaming.
// ---------------------------------------------------------------------------
template <int SELFMASK>
__global__ void __launch_bounds__(256)
ucl_wmma_gemm_lse(const _Float16* __restrict__ ZA, const _Float16* __restrict__ ZB,
                  float* __restrict__ S) {
  __shared__ __attribute__((aligned(32))) _Float16 panel[2][PANEL_COLS * 128];
  const int lane = threadIdx.x & 31;
  const int wave = threadIdx.x >> 5;
  const int row0 = blockIdx.x * 128 + wave * 16;
  // K = 128: keep all four A fragments resident
  v16h a0 = load_a_frag(ZA, 128, row0, 0,  lane);
  v16h a1 = load_a_frag(ZA, 128, row0, 32, lane);
  v16h a2 = load_a_frag(ZA, 128, row0, 64, lane);
  v16h a3 = load_a_frag(ZA, 128, row0, 96, lane);
  float accs[8];
#pragma unroll
  for (int i = 0; i < 8; ++i) accs[i] = 0.0f;
  const int rb = row0 + ((lane >> 4) << 3);
  const int n  = lane & 15;

  const int tid16 = threadIdx.x * 16;  // byte offset of this thread's slice
  // prologue: stage panel 0 into buffer 0
  {
    const char* g = (const char*)ZB + tid16;
    unsigned int l = (unsigned int)(uintptr_t)((char*)&panel[0][0] + tid16);
#pragma unroll
    for (int i = 0; i < 4; ++i) ucl_async_ld16(l + i * 4096, g + i * 4096);
  }

  const int NSUP = NROWS / PANEL_COLS;  // 128 super-tiles
  for (int st = 0; st < NSUP; ++st) {
    const int cur = st & 1;
    if (st + 1 < NSUP) {
      const char* g = (const char*)ZB + (size_t)(st + 1) * PANEL_BYTES + tid16;
      unsigned int l = (unsigned int)(uintptr_t)((char*)&panel[cur ^ 1][0] + tid16);
#pragma unroll
      for (int i = 0; i < 4; ++i) ucl_async_ld16(l + i * 4096, g + i * 4096);
      ucl_wait_async4();   // retire panel st's 4 copies (async loads complete in order)
    } else {
      ucl_wait_async0();
    }
    __syncthreads();
    const _Float16* lb = &panel[cur][0];
#pragma unroll
    for (int ct2 = 0; ct2 < 4; ++ct2) {
      v16h b0 = lds_b_frag(lb, ct2, 0,  lane);
      v16h b1 = lds_b_frag(lb, ct2, 32, lane);
      v16h b2 = lds_b_frag(lb, ct2, 64, lane);
      v16h b3 = lds_b_frag(lb, ct2, 96, lane);
      v8f c = v8f_zero();
      c = __builtin_amdgcn_wmma_f32_16x16x32_f16(false, a0, false, b0, (short)0, c, false, false);
      c = __builtin_amdgcn_wmma_f32_16x16x32_f16(false, a1, false, b1, (short)0, c, false, false);
      c = __builtin_amdgcn_wmma_f32_16x16x32_f16(false, a2, false, b2, (short)0, c, false, false);
      c = __builtin_amdgcn_wmma_f32_16x16x32_f16(false, a3, false, b3, (short)0, c, false, false);
      const int col = st * PANEL_COLS + ct2 * 16 + n;
#pragma unroll
      for (int i = 0; i < 8; ++i) {
        float e = __expf(2.0f * c[i] - 2.0f);
        if (SELFMASK && (((rb + i) & 255) == col)) e = 0.0f;  // fill_diagonal_ per CHUNK
        accs[i] += e;
      }
    }
    __syncthreads();
  }
  // reduce across the 16 lanes sharing each row (xor within half-wave)
#pragma unroll
  for (int i = 0; i < 8; ++i) {
    float v = accs[i];
    v += __shfl_xor(v, 1); v += __shfl_xor(v, 2);
    v += __shfl_xor(v, 4); v += __shfl_xor(v, 8);
    accs[i] = v;
  }
  if ((lane & 15) == 0) {
#pragma unroll
    for (int i = 0; i < 8; ++i) S[rb + i] = accs[i];
  }
}

// ---------------------------------------------------------------------------
// Final loss: warp per row. pos = dot(za,zb)/TEMP; lse = log(Scross+Sself)+2.
// ---------------------------------------------------------------------------
__global__ void ucl_zero_out(float* o) { if (threadIdx.x == 0 && blockIdx.x == 0) o[0] = 0.0f; }

__global__ void ucl_final_loss(const _Float16* __restrict__ Z0, const _Float16* __restrict__ Z1,
                               const _Float16* __restrict__ Z2,
                               const float* __restrict__ S01, const float* __restrict__ S02,
                               const float* __restrict__ S12, const float* __restrict__ S00,
                               const float* __restrict__ S11, float* __restrict__ out) {
  const int lane = threadIdx.x & 31;
  const int w    = threadIdx.x >> 5;
  const int row  = blockIdx.x * 8 + w;
  const size_t base = (size_t)row * 128;
  float d01 = 0.0f, d02 = 0.0f, d12 = 0.0f;
#pragma unroll
  for (int j = lane; j < 128; j += 32) {
    float a = (float)Z0[base + j], b = (float)Z1[base + j], c = (float)Z2[base + j];
    d01 += a * b; d02 += a * c; d12 += b * c;
  }
#pragma unroll
  for (int m = 1; m < 32; m <<= 1) {
    d01 += __shfl_xor(d01, m);
    d02 += __shfl_xor(d02, m);
    d12 += __shfl_xor(d12, m);
  }
  __shared__ float sh[8];
  if (lane == 0) {
    float t = (-2.0f * d01 + __logf(S01[row] + S00[row]) + 2.0f)
            + (-2.0f * d02 + __logf(S02[row] + S00[row]) + 2.0f)
            + (-2.0f * d12 + __logf(S12[row] + S11[row]) + 2.0f);
    sh[w] = t;
  }
  __syncthreads();
  if (threadIdx.x == 0) {
    float s = 0.0f;
#pragma unroll
    for (int i = 0; i < 8; ++i) s += sh[i];
    atomicAdd(out, s * (1.0f / (3.0f * (float)NROWS)));
  }
}

// ---------------------------------------------------------------------------
// Host driver
// ---------------------------------------------------------------------------
extern "C" void kernel_launch(void* const* d_in, const int* in_sizes, int n_in,
                              void* d_out, int out_size, void* d_ws, size_t ws_size,
                              hipStream_t stream) {
  const float* x0   = (const float*)d_in[0];
  const float* x1   = (const float*)d_in[1];
  const float* x2   = (const float*)d_in[2];
  const float* p_w1 = (const float*)d_in[3];
  const float* p_b1 = (const float*)d_in[4];
  const float* p_g  = (const float*)d_in[5];
  const float* p_bt = (const float*)d_in[6];
  const float* p_w2 = (const float*)d_in[7];
  const float* p_b2 = (const float*)d_in[8];
  const float* e_w1 = (const float*)d_in[9];
  const float* e_b1 = (const float*)d_in[10];
  const float* e_g  = (const float*)d_in[11];
  const float* e_bt = (const float*)d_in[12];
  const float* e_w2 = (const float*)d_in[13];
  const float* e_b2 = (const float*)d_in[14];
  (void)in_sizes; (void)n_in; (void)out_size; (void)ws_size;

  char* W = (char*)d_ws;
  size_t off = 0;
  auto A = [&](size_t b) { size_t r = off; off += (b + 255) & ~(size_t)255; return r; };
  const size_t zbytes = (size_t)NROWS * 128 * 2;
  const size_t oZ0 = A(zbytes), oZ1 = A(zbytes), oZ2 = A(zbytes);
  const size_t oS01 = A(NROWS * 4), oS02 = A(NROWS * 4), oS12 = A(NROWS * 4);
  const size_t oS00 = A(NROWS * 4), oS11 = A(NROWS * 4);
  const size_t oSC = A(512), oSH = A(512);
  const size_t oXH  = A((size_t)NROWS * 832 * 2);
  const size_t oW1T = A((size_t)128 * 832 * 2);
  const size_t oW2T = A((size_t)128 * 128 * 2);
  const size_t oH   = A((size_t)NROWS * 128 * 4);
  const size_t oHN  = A((size_t)NROWS * 128 * 2);
  const size_t oZT  = A((size_t)NROWS * 128 * 4);

  _Float16* Z0  = (_Float16*)(W + oZ0);
  _Float16* Z1  = (_Float16*)(W + oZ1);
  _Float16* Z2  = (_Float16*)(W + oZ2);
  float* S01 = (float*)(W + oS01); float* S02 = (float*)(W + oS02);
  float* S12 = (float*)(W + oS12); float* S00 = (float*)(W + oS00);
  float* S11 = (float*)(W + oS11);
  float* SC = (float*)(W + oSC); float* SH = (float*)(W + oSH);
  _Float16* XH  = (_Float16*)(W + oXH);
  _Float16* W1T = (_Float16*)(W + oW1T);
  _Float16* W2T = (_Float16*)(W + oW2T);
  float* Hbuf = (float*)(W + oH);
  _Float16* HN = (_Float16*)(W + oHN);
  float* ZT = (float*)(W + oZT);

  auto run_proj = [&](const float* x, int Kin, int Kp,
                      const float* w1, const float* b1, const float* g, const float* bt,
                      const float* w2, const float* b2, _Float16* Zout) {
    int tot1 = NROWS * Kp;
    ucl_cast_pad<<<(tot1 + 255) / 256, 256, 0, stream>>>(x, XH, Kin, Kp, tot1);
    int totw = 128 * Kp;
    ucl_wtrans<<<(totw + 255) / 256, 256, 0, stream>>>(w1, W1T, Kin, Kp, totw);
    ucl_wtrans<<<64, 256, 0, stream>>>(w2, W2T, 128, 128, 128 * 128);
    ucl_wmma_gemm_bias<<<64, 256, 0, stream>>>(XH, W1T, b1, Hbuf, Kp);
    ucl_col_stats<<<128, 256, 0, stream>>>(Hbuf, g, bt, SC, SH);
    ucl_bn_relu<<<(NROWS * 128) / 256, 256, 0, stream>>>(Hbuf, SC, SH, HN);
    ucl_wmma_gemm_bias<<<64, 256, 0, stream>>>(HN, W2T, b2, ZT, 128);
    ucl_row_norm<<<NROWS / 8, 256, 0, stream>>>(ZT, Zout);
  };

  run_proj(x0, 512, 512, p_w1, p_b1, p_g, p_bt, p_w2, p_b2, Z0);
  run_proj(x1, 812, 832, e_w1, e_b1, e_g, e_bt, e_w2, e_b2, Z1);
  run_proj(x2, 512, 512, p_w1, p_b1, p_g, p_bt, p_w2, p_b2, Z2);

  ucl_wmma_gemm_lse<0><<<64, 256, 0, stream>>>(Z0, Z1, S01);
  ucl_wmma_gemm_lse<0><<<64, 256, 0, stream>>>(Z0, Z2, S02);
  ucl_wmma_gemm_lse<0><<<64, 256, 0, stream>>>(Z1, Z2, S12);
  ucl_wmma_gemm_lse<1><<<64, 256, 0, stream>>>(Z0, Z0, S00);
  ucl_wmma_gemm_lse<1><<<64, 256, 0, stream>>>(Z1, Z1, S11);

  ucl_zero_out<<<1, 1, 0, stream>>>((float*)d_out);
  ucl_final_loss<<<NROWS / 8, 256, 0, stream>>>(Z0, Z1, Z2, S01, S02, S12, S00, S11,
                                                (float*)d_out);
}
